// ProteinFeatures_11673721110547
// MI455X (gfx1250) — compile-verified, matrix operand order checked by reading, not run on hardware
//
#include <hip/hip_runtime.h>
#include <hip/hip_bf16.h>
#include <math.h>

#define DEVINL __device__ __forceinline__

typedef __attribute__((ext_vector_type(16))) __bf16 v16bf;
typedef __attribute__((ext_vector_type(8)))  float  v8f;
typedef __attribute__((ext_vector_type(4)))  unsigned int v4u;
typedef __attribute__((ext_vector_type(8)))  int    v8i_;
typedef __attribute__((ext_vector_type(4)))  int    v4i_;

constexpr int B_ = 2, N_ = 4096, A_ = 5, TOPK_ = 30, KRBF_ = 16;
constexpr int MAXREL_ = 32, MAXAA_ = 22;
constexpr int EDGE_IN_ = 446;               // 16 + 400 + 14 + 16
constexpr int KPAD_ = 448;                  // padded K (14 * 32)
constexpr int NOUT_ = 128;
constexpr int M_TOTAL = B_ * N_ * TOPK_;    // 245760

// ---- workspace layout (bytes; every block 256-aligned) ----
constexpr size_t OFF_E   = 0;                                   // M*448 bf16
constexpr size_t OFF_W1T = OFF_E   + (size_t)M_TOTAL*KPAD_*2;   // 448*448 bf16, transposed [n][k]
constexpr size_t OFF_W2T = OFF_W1T + (size_t)KPAD_*KPAD_*2;     // 128*448 bf16, transposed [n][k]
constexpr size_t OFF_IDX = OFF_W2T + (size_t)NOUT_*KPAD_*2;     // M int32
constexpr size_t OFF_LF  = OFF_IDX + (size_t)M_TOTAL*4;         // B*N*9 float

DEVINL float fsign(float x) { return (x > 0.f) ? 1.f : ((x < 0.f) ? -1.f : 0.f); }
DEVINL float gelu_exact(float x) { return 0.5f * x * (1.f + erff(x * 0.70710678118654752f)); }
DEVINL __bf16 tobf(float x) { return (__bf16)x; }

// ---- CDNA5 async global->LDS copy (ASYNCcnt path, bypasses VGPRs) ----
// LDS offset: for shared-memory flat addresses, the LDS offset is the low 32 bits of the address (ISA 10.2)
DEVINL unsigned lds_off_of(const void* p) { return (unsigned)(unsigned long long)(uintptr_t)p; }

DEVINL void async_copy_b128(unsigned lds_off, unsigned long long gaddr)
{
    asm volatile("global_load_async_to_lds_b128 %0, %1, off"
                 :: "v"(lds_off), "v"(gaddr) : "memory");
}
DEVINL void wait_asynccnt0() { asm volatile("s_wait_asynccnt 0" ::: "memory"); }

// ---- CDNA5 Tensor Data Mover: 2D tile global->LDS (TENSORcnt path) ----
// Descriptor per ISA 8.3/8.4. data_size = 4B units; tile rows packed contiguous in LDS.
DEVINL void tdm_load_2d(unsigned lds_addr, const void* gptr,
                        unsigned tile_d0_units, unsigned tile_d1_rows,
                        unsigned long long d0_stride_units)
{
    unsigned long long ga = (unsigned long long)(uintptr_t)gptr;
    unsigned td0 = tile_d0_units, td1 = tile_d1_rows;   // tensor dims = tile dims (no OOB)
    v4u g0;
    g0[0] = 1u;                                          // count=1 (valid), flags 0
    g0[1] = lds_addr;                                    // lds_addr [63:32]
    g0[2] = (unsigned)(ga & 0xFFFFFFFFu);                // global_addr low
    g0[3] = (unsigned)((ga >> 32) & 0x01FFFFFFu) | (2u << 30);   // addr hi + type=2
    v8i_ g1;
    g1[0] = (int)(2u << 16);                             // wg_mask=0, data_size=2 (4B)
    g1[1] = (int)((td0 & 0xFFFFu) << 16);                // tensor_dim0[15:0] @ [79:64... ] (bits 79:48)
    g1[2] = (int)((td0 >> 16) | ((td1 & 0xFFFFu) << 16));// tensor_dim0 hi | tensor_dim1 lo
    g1[3] = (int)((td1 >> 16) | (tile_d0_units << 16));  // tensor_dim1 hi | tile_dim0
    g1[4] = (int)(tile_d1_rows & 0xFFFFu);               // tile_dim1; tile_dim2=0 (2D)
    g1[5] = (int)(d0_stride_units & 0xFFFFFFFFu);        // tensor_dim0_stride low
    g1[6] = (int)((d0_stride_units >> 32) & 0xFFFFu);    // stride hi; dim1_stride=0 (2D)
    g1[7] = 0;
    v4i_ z4 = {0, 0, 0, 0};
#if defined(__clang_major__) && (__clang_major__ >= 23)
    v8i_ z8 = {0, 0, 0, 0, 0, 0, 0, 0};
    __builtin_amdgcn_tensor_load_to_lds(g0, g1, z4, z4, z8, 0);
#else
    __builtin_amdgcn_tensor_load_to_lds(g0, g1, z4, z4, 0);
#endif
}

// =====================================================================
// Kernel 1: per-row kNN (TOPK smallest masked distances, ties -> lowest idx)
// =====================================================================
__global__ __launch_bounds__(256) void topk_kernel(const float* __restrict__ X,
                                                   const float* __restrict__ mask,
                                                   int* __restrict__ E_idx)
{
    __shared__ float Dsh[N_];
    __shared__ float rv[256];
    __shared__ int   ri[256];
    const int row = blockIdx.x;
    const int b   = row / N_;
    const int i   = row - b * N_;
    const int tid = threadIdx.x;
    const float* Xb = X + (size_t)b * N_ * (A_ * 3);
    const float cix = Xb[i*15+3], ciy = Xb[i*15+4], ciz = Xb[i*15+5];
    const float mi  = mask[b*N_+i];

    float lmax = 0.f;
    for (int j = tid; j < N_; j += 256) {
        float dx = Xb[j*15+3]-cix, dy = Xb[j*15+4]-ciy, dz = Xb[j*15+5]-ciz;
        float d  = sqrtf(dx*dx + dy*dy + dz*dz + 1e-6f);
        float m2 = mi * mask[b*N_+j];
        float dm = m2 * d;
        Dsh[j] = dm;
        lmax = fmaxf(lmax, dm);
    }
    rv[tid] = lmax;
    __syncthreads();
    for (int s = 128; s > 0; s >>= 1) {
        if (tid < s) rv[tid] = fmaxf(rv[tid], rv[tid+s]);
        __syncthreads();
    }
    const float rowmax = rv[0];
    __syncthreads();
    for (int j = tid; j < N_; j += 256) {
        float m2 = mi * mask[b*N_+j];
        Dsh[j] += (1.f - m2) * rowmax;
    }
    __syncthreads();

    for (int it = 0; it < TOPK_; ++it) {
        float bv = 3.4e38f; int bidx = 0x7fffffff;
        for (int j = tid; j < N_; j += 256) {
            float v = Dsh[j];
            if (v < bv) { bv = v; bidx = j; }   // ascending j keeps lowest index on ties
        }
        rv[tid] = bv; ri[tid] = bidx;
        __syncthreads();
        for (int s = 128; s > 0; s >>= 1) {
            if (tid < s) {
                float v2 = rv[tid+s]; int i2 = ri[tid+s];
                if (v2 < rv[tid] || (v2 == rv[tid] && i2 < ri[tid])) { rv[tid] = v2; ri[tid] = i2; }
            }
            __syncthreads();
        }
        if (tid == 0) { E_idx[row*TOPK_ + it] = ri[0]; Dsh[ri[0]] = 3.4e38f; }
        __syncthreads();
    }
}

// =====================================================================
// Kernel 2: per-node local frames lf = rows {b_, n_, b_ x n_}
// =====================================================================
DEVINL void l2n3(float& x, float& y, float& z) {
    float n = sqrtf(x*x + y*y + z*z);
    float inv = 1.f / fmaxf(n, 1e-12f);
    x *= inv; y *= inv; z *= inv;
}

__global__ __launch_bounds__(256) void frames_kernel(const float* __restrict__ X,
                                                     float* __restrict__ lf)
{
    int idx = blockIdx.x * 256 + threadIdx.x;
    if (idx >= B_ * N_) return;
    int b = idx / N_;
    int n = idx - b * N_;
    const float* Xb = X + (size_t)b * N_ * 15;
    auto ca = [&](int m, float v[3]) { v[0]=Xb[m*15+3]; v[1]=Xb[m*15+4]; v[2]=Xb[m*15+5]; };

    float u0[3], u1[3];
    if (n == 0) { u0[0]=u0[1]=u0[2]=1.f; }
    else { float a[3], c[3]; ca(n, a); ca(n-1, c); u0[0]=a[0]-c[0]; u0[1]=a[1]-c[1]; u0[2]=a[2]-c[2]; }
    l2n3(u0[0], u0[1], u0[2]);

    float bx, by, bz, nx, ny, nz;
    if (n < N_ - 1) {
        float a[3], c[3]; ca(n+1, a); ca(n, c);
        u1[0]=a[0]-c[0]; u1[1]=a[1]-c[1]; u1[2]=a[2]-c[2];
        l2n3(u1[0], u1[1], u1[2]);
        bx = u0[0]-u1[0]; by = u0[1]-u1[1]; bz = u0[2]-u1[2];
        nx = u0[1]*u1[2]-u0[2]*u1[1];
        ny = u0[2]*u1[0]-u0[0]*u1[2];
        nz = u0[0]*u1[1]-u0[1]*u1[0];
    } else {
        bx = by = bz = 1.f; nx = ny = nz = 1.f;
    }
    l2n3(bx, by, bz);
    l2n3(nx, ny, nz);
    float cx = by*nz - bz*ny, cy = bz*nx - bx*nz, cz = bx*ny - by*nx;  // not normalized (matches ref)
    float* o = lf + (size_t)idx * 9;
    o[0]=bx; o[1]=by; o[2]=bz; o[3]=nx; o[4]=ny; o[5]=nz; o[6]=cx; o[7]=cy; o[8]=cz;
}

// =====================================================================
// Kernel 3a/3b: weight conversion to transposed, padded bf16
// =====================================================================
__global__ __launch_bounds__(256) void convert_w1(const float* __restrict__ W1, __bf16* __restrict__ W1t)
{
    int idx = blockIdx.x * 256 + threadIdx.x;
    if (idx >= KPAD_ * KPAD_) return;
    int n = idx / KPAD_, k = idx - n * KPAD_;
    float v = (n < EDGE_IN_ && k < EDGE_IN_) ? W1[(size_t)k * EDGE_IN_ + n] : 0.f;
    W1t[idx] = tobf(v);
}
__global__ __launch_bounds__(256) void convert_w2(const float* __restrict__ W2, __bf16* __restrict__ W2t)
{
    int idx = blockIdx.x * 256 + threadIdx.x;
    if (idx >= NOUT_ * KPAD_) return;
    int n = idx / KPAD_, k = idx - n * KPAD_;
    float v = (k < EDGE_IN_) ? W2[(size_t)k * NOUT_ + n] : 0.f;
    W2t[idx] = tobf(v);
}

// =====================================================================
// Kernel 4: edge features -> E bf16 [M][448]
// 64 threads per edge, 4 edges per 256-thread block
// =====================================================================
__global__ __launch_bounds__(256) void edge_feat_kernel(
    const float* __restrict__ X, const float* __restrict__ mask,
    const float* __restrict__ mask_atoms, const float* __restrict__ means,
    const float* __restrict__ stds, const float* __restrict__ mul_w,
    const float* __restrict__ bias_w, const float* __restrict__ aa_pair_embed,
    const float* __restrict__ relpos_embed, const float* __restrict__ mask_embed,
    const int* __restrict__ aa, const int* __restrict__ residue_idx,
    const int* __restrict__ chain_labels, const int* __restrict__ E_idx,
    const float* __restrict__ lf, __bf16* __restrict__ E)
{
    __shared__ float s_xgm[4][25];
    __shared__ float s_tq[4][8];

    const int tid = threadIdx.x;
    const int eo  = tid >> 6;        // edge slot in block
    const int lt  = tid & 63;        // lane within edge group
    const int e   = blockIdx.x * 4 + eo;

    const int b   = e / (N_ * TOPK_);
    const int rem = e - b * (N_ * TOPK_);
    const int i   = rem / TOPK_;
    const int j   = E_idx[e];

    const float ma = mask[b*N_+i] * mask[b*N_+j];
    const float* Xi = X + ((size_t)(b*N_+i)) * 15;
    const float* Xj = X + ((size_t)(b*N_+j)) * 15;

    if (lt < 25) {                    // atom-pair distances + mul/bias + atom mask
        int anb = lt / 5, asf = lt - anb * 5;
        float dx = Xj[anb*3+0] - Xi[asf*3+0];
        float dy = Xj[anb*3+1] - Xi[asf*3+1];
        float dz = Xj[anb*3+2] - Xi[asf*3+2];
        float dab = sqrtf(dx*dx + dy*dy + dz*dz + 1e-12f);
        float mf  = mask_atoms[(b*N_+j)*A_ + asf] * mask_atoms[(b*N_+j)*A_ + anb];
        s_xgm[eo][lt] = (mul_w[lt] * dab + bias_w[lt]) * mf;
    }
    if (lt == 32) {                   // t and quaternion (one lane of second wave)
        const float* li = lf + (size_t)(b*N_+i) * 9;
        const float* lj = lf + (size_t)(b*N_+j) * 9;
        float d0 = Xj[3]-Xi[3], d1 = Xj[4]-Xi[4], d2 = Xj[5]-Xi[5];
        float t0 = li[0]*d0 + li[1]*d1 + li[2]*d2;
        float t1 = li[3]*d0 + li[4]*d1 + li[5]*d2;
        float t2 = li[6]*d0 + li[7]*d1 + li[8]*d2;
        float tn = sqrtf(t0*t0 + t1*t1 + t2*t2);
        float ti = ma / fmaxf(tn, 1e-12f);
        t0 *= ti; t1 *= ti; t2 *= ti;
        // r = lf_i^T * lf_j
        float r[3][3];
        #pragma unroll
        for (int a = 0; a < 3; ++a)
            #pragma unroll
            for (int c = 0; c < 3; ++c)
                r[a][c] = li[0*3+a]*lj[0*3+c] + li[1*3+a]*lj[1*3+c] + li[2*3+a]*lj[2*3+c];
        float Rxx = r[0][0], Ryy = r[1][1], Rzz = r[2][2];
        float mx = 0.5f * sqrtf(fabsf(1.f + Rxx - Ryy - Rzz) + 1e-12f);
        float my = 0.5f * sqrtf(fabsf(1.f - Rxx + Ryy - Rzz) + 1e-12f);
        float mz = 0.5f * sqrtf(fabsf(1.f - Rxx - Ryy + Rzz) + 1e-12f);
        float qx = fsign(r[2][1] - r[1][2]) * mx;
        float qy = fsign(r[0][2] - r[2][0]) * my;
        float qz = fsign(r[1][0] - r[0][1]) * mz;
        float qw = sqrtf(fmaxf(1.f + Rxx + Ryy + Rzz, 0.f) + 1e-12f) * 0.5f;
        float qn = sqrtf(qx*qx + qy*qy + qz*qz + qw*qw);
        float qi = ma / fmaxf(qn, 1e-12f);
        s_tq[eo][0] = t0; s_tq[eo][1] = t1; s_tq[eo][2] = t2;
        s_tq[eo][3] = qx*qi; s_tq[eo][4] = qy*qi; s_tq[eo][5] = qz*qi; s_tq[eo][6] = qw*qi;
    }
    __syncthreads();

    __bf16* Er = E + (size_t)e * KPAD_;
    #pragma unroll
    for (int f = lt; f < KPAD_; f += 64) {
        float val;
        if (f < 16) {
            int off = residue_idx[b*N_+i] - residue_idx[b*N_+j];
            int d = min(max(off + MAXREL_, 0), 2*MAXREL_);
            int eq = (chain_labels[b*N_+i] == chain_labels[b*N_+j]) ? 1 : 0;
            val = relpos_embed[d*16 + f] + mask_embed[eq*16 + f];
        } else if (f < 416) {
            int p  = (f - 16) >> 4;
            int rr = (f - 16) & 15;
            float std = fabsf(stds[rr]) + 0.01f;
            float z = (s_xgm[eo][p] - means[rr]) / std;
            val = __expf(-0.5f * z * z) * (0.3989422804014327f / std) * ma;
        } else if (f < 430) {
            int o = f - 416;
            if (o < 7)       val = s_tq[eo][o];              // t then q
            else             val = 1.f - 2.f * fabsf(s_tq[eo][o-7]);
        } else if (f < 446) {
            int pair = aa[b*N_+i] * MAXAA_ + aa[b*N_+j];
            val = aa_pair_embed[pair*16 + (f - 430)] * ma;
        } else {
            val = 0.f;                                       // K padding
        }
        Er[f] = tobf(val);
    }
}

// =====================================================================
// Kernel 5: fused MLP  out = (gelu(E @ W1 + b1)) @ W2 + b2, all via WMMA
// Block = 256 threads (8 waves), 32 edge rows per block.
// Stage 1 staging via TDM tensor_load_to_lds (TENSORcnt);
// Stage 2 staging via GLOBAL_LOAD_ASYNC_TO_LDS_B128 (ASYNCcnt).
// =====================================================================
union FragBF { uint4 u4[2]; v16bf v; };

DEVINL v16bf load_frag(const __bf16* rowp, int half)
{
    FragBF f;
    f.u4[0] = *(const uint4*)(rowp + half * 8);        // K 0..7   or 8..15
    f.u4[1] = *(const uint4*)(rowp + 16 + half * 8);   // K 16..23 or 24..31
    return f.v;
}

__global__ __launch_bounds__(256, 1) void fused_mlp_kernel(
    const __bf16* __restrict__ E, const __bf16* __restrict__ W1t,
    const __bf16* __restrict__ W2t, const float* __restrict__ b1,
    const float* __restrict__ b2, float* __restrict__ out)
{
    // region0: [Atile 32x32 | Btile 448x32] bf16 (30720 B), reused as Hlds 32x448 (28672 B)
    // region1: W2tile 128x32 bf16 (8192 B)
    __shared__ __align__(16) unsigned char smem[30720 + 8192];
    __bf16* Atile  = (__bf16*)smem;
    __bf16* Btile  = (__bf16*)(smem + 2048);
    __bf16* Hlds   = (__bf16*)smem;
    __bf16* W2tile = (__bf16*)(smem + 30720);

    const unsigned ldsA  = lds_off_of(Atile);
    const unsigned ldsB  = lds_off_of(Btile);
    const unsigned ldsW2 = lds_off_of(W2tile);

    const int tid  = threadIdx.x;
    const int wave = tid >> 5;
    const int lane = tid & 31;
    const int half = lane >> 4;
    const int lcol = lane & 15;
    const int m0   = blockIdx.x * 32;

    // ---------------- Stage 1: H = gelu(E @ W1 + b1) ----------------
    const int wm1 = wave & 1;    // 2 row groups of 16
    const int wn1 = wave >> 1;   // 4 col groups of 112 (7 tiles)
    v8f acc1[7] = {};

    for (int kk = 0; kk < KPAD_; kk += 32) {
        if (wave == 0) {
            // B tile: 448 rows x 64B (16 dwords), row stride 896B (224 dwords)
            tdm_load_2d(ldsB, (const void*)(W1t + kk), 16u, 448u, 224ull);
            // A tile: 32 rows x 64B, row stride 896B
            tdm_load_2d(ldsA, (const void*)(E + (size_t)m0 * KPAD_ + kk), 16u, 32u, 224ull);
            __builtin_amdgcn_s_wait_tensorcnt(0);
        }
        __syncthreads();

        v16bf af = load_frag(Atile + (wm1*16 + lcol) * 32, half);
        v16bf bfr[7];
        #pragma unroll
        for (int t7 = 0; t7 < 7; ++t7)
            bfr[t7] = load_frag(Btile + (wn1 * 112 + t7 * 16 + lcol) * 32, half);
        #pragma unroll
        for (int t7 = 0; t7 < 7; ++t7)
            acc1[t7] = __builtin_amdgcn_wmma_f32_16x16x32_bf16(
                false, af, false, bfr[t7], (short)0, acc1[t7], false, false);
        __syncthreads();
    }

    // bias + exact GeLU, write hidden tile to LDS (bf16); padded cols are 0 -> gelu(0)=0
    #pragma unroll
    for (int t7 = 0; t7 < 7; ++t7) {
        int col = wn1 * 112 + t7 * 16 + lcol;
        float bv = (col < EDGE_IN_) ? b1[col] : 0.f;
        #pragma unroll
        for (int r = 0; r < 8; ++r) {
            int row = wm1 * 16 + half * 8 + r;
            float v = acc1[t7][r] + bv;
            Hlds[row * KPAD_ + col] = tobf(gelu_exact(v));
        }
    }
    __syncthreads();

    // ---------------- Stage 2: out = H @ W2 + b2 ----------------
    const int wm2 = wave & 1;    // 2 row groups of 16
    const int wn2 = wave >> 1;   // 4 col groups of 32 (2 tiles)
    v8f acc2[2] = {};

    for (int kk = 0; kk < KPAD_; kk += 32) {
        #pragma unroll
        for (int q = 0; q < 2; ++q) {
            int idx = tid + q * 256;
            int row = idx >> 2, seg = idx & 3;
            async_copy_b128(ldsW2 + (unsigned)(row*32 + seg*8)*2,
                            (unsigned long long)(uintptr_t)(W2t + (size_t)row * KPAD_ + kk + seg*8));
        }
        wait_asynccnt0();
        __syncthreads();

        v16bf af = load_frag(Hlds + (wm2*16 + lcol) * KPAD_ + kk, half);
        v16bf bf0 = load_frag(W2tile + (wn2 * 32 + 0 * 16 + lcol) * 32, half);
        v16bf bf1 = load_frag(W2tile + (wn2 * 32 + 1 * 16 + lcol) * 32, half);
        acc2[0] = __builtin_amdgcn_wmma_f32_16x16x32_bf16(
            false, af, false, bf0, (short)0, acc2[0], false, false);
        acc2[1] = __builtin_amdgcn_wmma_f32_16x16x32_bf16(
            false, af, false, bf1, (short)0, acc2[1], false, false);
        __syncthreads();
    }

    #pragma unroll
    for (int t2 = 0; t2 < 2; ++t2) {
        int col = wn2 * 32 + t2 * 16 + lcol;
        float bv = b2[col];
        #pragma unroll
        for (int r = 0; r < 8; ++r) {
            int row = m0 + wm2 * 16 + half * 8 + r;
            out[(size_t)row * NOUT_ + col] = acc2[t2][r] + bv;
        }
    }
}

// =====================================================================
extern "C" void kernel_launch(void* const* d_in, const int* in_sizes, int n_in,
                              void* d_out, int out_size, void* d_ws, size_t ws_size,
                              hipStream_t stream)
{
    const float* X             = (const float*)d_in[0];
    const float* mask          = (const float*)d_in[1];
    const float* mask_atoms    = (const float*)d_in[2];
    const float* means         = (const float*)d_in[3];
    const float* stds          = (const float*)d_in[4];
    const float* mul_w         = (const float*)d_in[5];
    const float* bias_w        = (const float*)d_in[6];
    const float* aa_pair_embed = (const float*)d_in[7];
    const float* relpos_embed  = (const float*)d_in[8];
    const float* mask_embed    = (const float*)d_in[9];
    const float* W1            = (const float*)d_in[10];
    const float* b1            = (const float*)d_in[11];
    const float* W2            = (const float*)d_in[12];
    const float* b2            = (const float*)d_in[13];
    const int*   aa            = (const int*)d_in[14];
    const int*   residue_idx   = (const int*)d_in[15];
    const int*   chain_labels  = (const int*)d_in[16];
    float* out = (float*)d_out;

    char* ws = (char*)d_ws;
    __bf16* Ebf  = (__bf16*)(ws + OFF_E);
    __bf16* W1t  = (__bf16*)(ws + OFF_W1T);
    __bf16* W2t  = (__bf16*)(ws + OFF_W2T);
    int*    Eidx = (int*)   (ws + OFF_IDX);
    float*  lfbuf= (float*) (ws + OFF_LF);

    topk_kernel<<<B_ * N_, 256, 0, stream>>>(X, mask, Eidx);
    frames_kernel<<<(B_ * N_ + 255) / 256, 256, 0, stream>>>(X, lfbuf);
    convert_w1<<<(KPAD_ * KPAD_ + 255) / 256, 256, 0, stream>>>(W1, W1t);
    convert_w2<<<(NOUT_ * KPAD_ + 255) / 256, 256, 0, stream>>>(W2, W2t);
    edge_feat_kernel<<<M_TOTAL / 4, 256, 0, stream>>>(
        X, mask, mask_atoms, means, stds, mul_w, bias_w, aa_pair_embed,
        relpos_embed, mask_embed, aa, residue_idx, chain_labels, Eidx, lfbuf, Ebf);
    fused_mlp_kernel<<<M_TOTAL / 32, 256, 0, stream>>>(Ebf, W1t, W2t, b1, b2, out);
}
